// NATBlock_63745904607567
// MI455X (gfx1250) — matrix operand order repaired and checked by reading, not compiled, not run on hardware
//
#include <hip/hip_runtime.h>
#include <hip/hip_bf16.h>

typedef __attribute__((ext_vector_type(16))) _Float16 v16h;
typedef __attribute__((ext_vector_type(8)))  _Float16 v8h;
typedef __attribute__((ext_vector_type(8)))  float    v8f;

#define BN 8
#define LN_ 4096
#define CN 256
#define HN 8
#define HDN 32
#define KWIN 7
#define NAT_SCALE 0.17677669529663687f   // 32^-0.5

// ---------------------------------------------------------------------------
// Weight convert: W (K x N, f32 row-major)  ->  Wt (N x K, f16 row-major)
// ---------------------------------------------------------------------------
__global__ void convert_transpose_kernel(const float* __restrict__ W,
                                         _Float16* __restrict__ Wt,
                                         int K, int N) {
  int i = blockIdx.x * 256 + threadIdx.x;
  if (i >= N * K) return;
  int n = i / K;
  int k = i - n * K;
  Wt[i] = (_Float16)W[(size_t)k * N + n];
}

// ds_w (512,256,3) OIH -> Wt (512, 768) with Wt[co][kk*256+ci] = W[co][ci][kk]
__global__ void convert_dsw_kernel(const float* __restrict__ W,
                                   _Float16* __restrict__ Wt) {
  int i = blockIdx.x * 256 + threadIdx.x;
  if (i >= 512 * 768) return;
  int co = i / 768;
  int r  = i - co * 768;
  int kk = r >> 8;
  int ci = r & 255;
  Wt[i] = (_Float16)W[(size_t)co * 768 + ci * 3 + kk];
}

__global__ void copy_f32_kernel(const float* __restrict__ src,
                                float* __restrict__ dst, int n) {
  int i = blockIdx.x * 256 + threadIdx.x;
  if (i < n) dst[i] = src[i];
}

// ---------------------------------------------------------------------------
// LayerNorm over a row of length n (n == blockDim.x, power of two).
// Writes f16 (for WMMA input) and/or f32.
// ---------------------------------------------------------------------------
__global__ __launch_bounds__(512)
void ln_kernel(const float* __restrict__ x, const float* __restrict__ w,
               const float* __restrict__ b, _Float16* __restrict__ outH,
               float* __restrict__ outF, int n) {
  __shared__ float s1[512];
  __shared__ float s2[512];
  int row = blockIdx.x;
  int t = threadIdx.x;
  float v = x[(size_t)row * n + t];
  s1[t] = v;
  s2[t] = v * v;
  __syncthreads();
  for (int st = n >> 1; st > 0; st >>= 1) {
    if (t < st) { s1[t] += s1[t + st]; s2[t] += s2[t + st]; }
    __syncthreads();
  }
  float m   = s1[0] / n;
  float var = s2[0] / n - m * m;
  float y = (v - m) * rsqrtf(var + 1e-5f) * w[t] + b[t];
  if (outH) outH[(size_t)row * n + t] = (_Float16)y;
  if (outF) outF[(size_t)row * n + t] = y;
}

// ---------------------------------------------------------------------------
// WMMA GEMM: C(MxN) = A(MxK,f16) * Bt(NxK,f16)^T  with fused epilogues.
// 256 thr (8 waves), block tile 128x128, wave tile 64x32, K-step 32.
// Double-buffered LDS: global loads for tile t+1 issue before the 8 WMMAs of
// tile t (compute hides HBM latency); one barrier per K-step.
// ---------------------------------------------------------------------------
#define EPI_F32        0
#define EPI_F16        1
#define EPI_GELU_F16   2
#define EPI_RESADD_F32 3

template <int EPI>
__global__ __launch_bounds__(256)
void wmma_gemm_kernel(const _Float16* __restrict__ A,
                      const _Float16* __restrict__ Bt,
                      const float* __restrict__ bias,
                      const float* __restrict__ res,
                      float* __restrict__ outF,
                      _Float16* __restrict__ outH,
                      int M, int N, int K) {
  __shared__ _Float16 lA[2][128][40];   // padded: 80B rows keep 16B alignment
  __shared__ _Float16 lB[2][128][40];

  const int tid   = threadIdx.x;
  const int lane  = tid & 31;
  const int wave  = tid >> 5;
  const int wm    = wave >> 2;          // 0..1  (M direction, 64 each)
  const int wn    = wave & 3;           // 0..3  (N direction, 32 each)
  const int halfw = lane >> 4;          // 0/1
  const int mr    = lane & 15;

  const int blockM = blockIdx.y * 128;
  const int blockN = blockIdx.x * 128;

  const int ldRow = tid >> 1;           // 0..127
  const int ldSeg = tid & 1;            // 0..1 (16 f16 each)

  const _Float16* gA = A  + (size_t)(blockM + ldRow) * K + ldSeg * 16;
  const _Float16* gB = Bt + (size_t)(blockN + ldRow) * K + ldSeg * 16;

  v8f acc[4][2] = {};

  const int nk = K >> 5;

  // prologue: stage tile 0 into buffer 0
  {
    v8h a0 = *(const v8h*)(gA);
    v8h a1 = *(const v8h*)(gA + 8);
    v8h b0 = *(const v8h*)(gB);
    v8h b1 = *(const v8h*)(gB + 8);
    *(v8h*)&lA[0][ldRow][ldSeg * 16]     = a0;
    *(v8h*)&lA[0][ldRow][ldSeg * 16 + 8] = a1;
    *(v8h*)&lB[0][ldRow][ldSeg * 16]     = b0;
    *(v8h*)&lB[0][ldRow][ldSeg * 16 + 8] = b1;
  }
  __syncthreads();

  int cur = 0;
  for (int t = 0; t < nk; ++t) {
    // issue global loads for next K tile (last iter: clamped, harmless reload)
    int ktn = (t + 1 < nk) ? (t + 1) << 5 : t << 5;
    v8h a0 = *(const v8h*)(gA + ktn);
    v8h a1 = *(const v8h*)(gA + ktn + 8);
    v8h b0 = *(const v8h*)(gB + ktn);
    v8h b1 = *(const v8h*)(gB + ktn + 8);

    // fragment loads from current buffer: issue ALL before any WMMA
    v16h bf[2];
#pragma unroll
    for (int tn = 0; tn < 2; ++tn) {
      const _Float16* pb = &lB[cur][wn * 32 + tn * 16 + mr][halfw * 16];
      v8h lo = *(const v8h*)pb;
      v8h hi = *(const v8h*)(pb + 8);
      bf[tn] = __builtin_shufflevector(lo, hi, 0, 1, 2, 3, 4, 5, 6, 7,
                                       8, 9, 10, 11, 12, 13, 14, 15);
    }
    v16h af[4];
#pragma unroll
    for (int tm = 0; tm < 4; ++tm) {
      const _Float16* pa = &lA[cur][wm * 64 + tm * 16 + mr][halfw * 16];
      v8h lo = *(const v8h*)pa;
      v8h hi = *(const v8h*)(pa + 8);
      af[tm] = __builtin_shufflevector(lo, hi, 0, 1, 2, 3, 4, 5, 6, 7,
                                       8, 9, 10, 11, 12, 13, 14, 15);
    }
    // 8 back-to-back WMMAs (XDL) while next tile's global loads are in flight
#pragma unroll
    for (int tm = 0; tm < 4; ++tm)
#pragma unroll
      for (int tn = 0; tn < 2; ++tn)
        acc[tm][tn] = __builtin_amdgcn_wmma_f32_16x16x32_f16(
            false, af[tm], false, bf[tn], (short)0, acc[tm][tn], false, false);

    // stage next tile into the other buffer
    int nxt = cur ^ 1;
    *(v8h*)&lA[nxt][ldRow][ldSeg * 16]     = a0;
    *(v8h*)&lA[nxt][ldRow][ldSeg * 16 + 8] = a1;
    *(v8h*)&lB[nxt][ldRow][ldSeg * 16]     = b0;
    *(v8h*)&lB[nxt][ldRow][ldSeg * 16 + 8] = b1;
    __syncthreads();
    cur = nxt;
  }

  // epilogue: VGPR r of lane -> row = tm*16 + halfw*8 + r, col = tn*16 + mr
  const int rowBase = blockM + wm * 64;
  const int colBase = blockN + wn * 32;
#pragma unroll
  for (int tm = 0; tm < 4; ++tm) {
#pragma unroll
    for (int tn = 0; tn < 2; ++tn) {
      int col = colBase + tn * 16 + mr;
      float bv = (EPI == EPI_F32) ? (bias ? bias[col] : 0.f) : bias[col];
#pragma unroll
      for (int r = 0; r < 8; ++r) {
        int row = rowBase + tm * 16 + halfw * 8 + r;
        size_t o = (size_t)row * N + col;
        float v = acc[tm][tn][r] + bv;
        if (EPI == EPI_F32) {
          outF[o] = v;
        } else if (EPI == EPI_F16) {
          outH[o] = (_Float16)v;
        } else if (EPI == EPI_GELU_F16) {
          float g = 0.5f * v * (1.0f + erff(v * 0.70710678118654752f));
          outH[o] = (_Float16)g;
        } else { // EPI_RESADD_F32
          outF[o] = res[o] + v;
        }
      }
    }
  }
}

// ---------------------------------------------------------------------------
// Neighborhood attention: one wave per (b,l,h); lane == head-dim (HD=32).
// qkv is f16 (B*L, 768) with col = t*256 + h*32 + d. Output f16 (B*L, 256).
// ---------------------------------------------------------------------------
__global__ __launch_bounds__(256)
void nat_attn_kernel(const _Float16* __restrict__ qkv,
                     const float* __restrict__ rpb,
                     _Float16* __restrict__ outH) {
  int w    = blockIdx.x * 8 + (threadIdx.x >> 5);
  int lane = threadIdx.x & 31;
  int h  = w & 7;
  int bl = w >> 3;                 // b*L + l
  int l  = bl & (LN_ - 1);

  const _Float16* base = qkv + (size_t)bl * 768;
  float q = (float)base[h * 32 + lane] * NAT_SCALE;

  int s = l - (KWIN / 2);
  if (s < 0) s = 0;
  if (s > LN_ - KWIN) s = LN_ - KWIN;
  int d0 = s - l;                  // window offset relative to l

  float logits[KWIN];
#pragma unroll
  for (int k = 0; k < KWIN; ++k) {
    const _Float16* kb = qkv + (size_t)(bl + d0 + k) * 768 + 256 + h * 32;
    float p = q * (float)kb[lane];
#pragma unroll
    for (int off = 16; off > 0; off >>= 1) p += __shfl_xor(p, off, 32);
    logits[k] = p + rpb[h * (2 * KWIN - 1) + (d0 + k + KWIN - 1)];
  }
  float mx = logits[0];
#pragma unroll
  for (int k = 1; k < KWIN; ++k) mx = fmaxf(mx, logits[k]);
  float e[KWIN], denom = 0.f;
#pragma unroll
  for (int k = 0; k < KWIN; ++k) { e[k] = __expf(logits[k] - mx); denom += e[k]; }
  float inv = 1.0f / denom;
  float o = 0.f;
#pragma unroll
  for (int k = 0; k < KWIN; ++k) {
    const _Float16* vb = qkv + (size_t)(bl + d0 + k) * 768 + 512 + h * 32;
    o += (e[k] * inv) * (float)vb[lane];
  }
  outH[(size_t)bl * 256 + h * 32 + lane] = (_Float16)o;
}

// ---------------------------------------------------------------------------
// im2col for stride-2 K=3 conv: out[(b*2048+lo)][kk*256+ci] = x[b][2lo+kk-1][ci]
// ---------------------------------------------------------------------------
__global__ void im2col_kernel(const float* __restrict__ x,
                              _Float16* __restrict__ out) {
  int row = blockIdx.x;            // b*2048 + lo
  int b  = row >> 11;
  int lo = row & 2047;
  int ci = threadIdx.x;            // 256
#pragma unroll
  for (int kk = 0; kk < 3; ++kk) {
    int li = 2 * lo + kk - 1;
    float v = (li >= 0 && li < LN_) ? x[((size_t)(b * LN_ + li)) * CN + ci] : 0.f;
    out[(size_t)row * 768 + kk * 256 + ci] = (_Float16)v;
  }
}

// ---------------------------------------------------------------------------
extern "C" void kernel_launch(void* const* d_in, const int* in_sizes, int n_in,
                              void* d_out, int out_size, void* d_ws, size_t ws_size,
                              hipStream_t stream) {
  (void)in_sizes; (void)n_in; (void)out_size; (void)ws_size;
  const float* x_in   = (const float*)d_in[0];
  const float* ln1_w  = (const float*)d_in[1];
  const float* ln1_b  = (const float*)d_in[2];
  const float* qkv_w  = (const float*)d_in[3];
  const float* qkv_b  = (const float*)d_in[4];
  const float* rpb    = (const float*)d_in[5];
  const float* proj_w = (const float*)d_in[6];
  const float* proj_b = (const float*)d_in[7];
  const float* ln2_w  = (const float*)d_in[8];
  const float* ln2_b  = (const float*)d_in[9];
  const float* fc1_w  = (const float*)d_in[10];
  const float* fc1_b  = (const float*)d_in[11];
  const float* fc2_w  = (const float*)d_in[12];
  const float* fc2_b  = (const float*)d_in[13];
  const float* ds_w   = (const float*)d_in[14];
  const float* dsln_w = (const float*)d_in[15];
  const float* dsln_b = (const float*)d_in[16];
  float* out = (float*)d_out;

  const int M    = BN * LN_;        // 32768 token rows
  const int Mds  = BN * 2048;       // 16384 conv rows
  const int ySz  = Mds * 512;       // 8388608
  const int xSz  = M * CN;          // 8388608

  // --- workspace carve-up ---
  char* ws = (char*)d_ws;
  size_t off = 0;
  auto take = [&](size_t bytes) -> void* {
    void* p = ws + off;
    off += (bytes + 255) & ~(size_t)255;
    return p;
  };
  _Float16* w_qkv[3]; _Float16* w_proj[3]; _Float16* w_fc1[3]; _Float16* w_fc2[3];
  for (int i = 0; i < 3; ++i) {
    w_qkv[i]  = (_Float16*)take((size_t)768 * 256 * 2);
    w_proj[i] = (_Float16*)take((size_t)256 * 256 * 2);
    w_fc1[i]  = (_Float16*)take((size_t)1024 * 256 * 2);
    w_fc2[i]  = (_Float16*)take((size_t)256 * 1024 * 2);
  }
  _Float16* w_ds   = (_Float16*)take((size_t)512 * 768 * 2);
  float*    xbuf   = (float*)   take((size_t)xSz * 4);
  _Float16* a16    = (_Float16*)take((size_t)M * CN * 2);
  _Float16* qkv16  = (_Float16*)take((size_t)M * 768 * 2);
  _Float16* h4_16  = (_Float16*)take((size_t)M * 1024 * 2);
  _Float16* col16  = (_Float16*)take((size_t)Mds * 768 * 2);
  float*    ypre   = (float*)   take((size_t)ySz * 4);

  // --- weights -> f16 (transposed to NxK) ---
  for (int i = 0; i < 3; ++i) {
    convert_transpose_kernel<<<(768 * 256 + 255) / 256, 256, 0, stream>>>(
        qkv_w + (size_t)i * 256 * 768, w_qkv[i], 256, 768);
    convert_transpose_kernel<<<(256 * 256 + 255) / 256, 256, 0, stream>>>(
        proj_w + (size_t)i * 256 * 256, w_proj[i], 256, 256);
    convert_transpose_kernel<<<(1024 * 256 + 255) / 256, 256, 0, stream>>>(
        fc1_w + (size_t)i * 256 * 1024, w_fc1[i], 256, 1024);
    convert_transpose_kernel<<<(256 * 1024 + 255) / 256, 256, 0, stream>>>(
        fc2_w + (size_t)i * 1024 * 256, w_fc2[i], 1024, 256);
  }
  convert_dsw_kernel<<<(512 * 768 + 255) / 256, 256, 0, stream>>>(ds_w, w_ds);

  // residual stream
  copy_f32_kernel<<<(xSz + 255) / 256, 256, 0, stream>>>(x_in, xbuf, xSz);

  for (int i = 0; i < 3; ++i) {
    // h = LN1(x) -> f16
    ln_kernel<<<M, 256, 0, stream>>>(xbuf, ln1_w + i * 256, ln1_b + i * 256,
                                     a16, nullptr, 256);
    // qkv = h @ Wqkv + b -> f16
    wmma_gemm_kernel<EPI_F16><<<dim3(768 / 128, M / 128), 256, 0, stream>>>(
        a16, w_qkv[i], qkv_b + i * 768, nullptr, nullptr, qkv16, M, 768, 256);
    // neighborhood attention -> f16
    nat_attn_kernel<<<(M * HN) / 8, 256, 0, stream>>>(
        qkv16, rpb + i * HN * (2 * KWIN - 1), a16);
    // x = x + attn @ Wproj + b
    wmma_gemm_kernel<EPI_RESADD_F32><<<dim3(256 / 128, M / 128), 256, 0, stream>>>(
        a16, w_proj[i], proj_b + i * 256, xbuf, xbuf, nullptr, M, 256, 256);
    // h = LN2(x) -> f16
    ln_kernel<<<M, 256, 0, stream>>>(xbuf, ln2_w + i * 256, ln2_b + i * 256,
                                     a16, nullptr, 256);
    // h4 = gelu(h @ Wfc1 + b) -> f16
    wmma_gemm_kernel<EPI_GELU_F16><<<dim3(1024 / 128, M / 128), 256, 0, stream>>>(
        a16, w_fc1[i], fc1_b + i * 1024, nullptr, nullptr, h4_16, M, 1024, 256);
    // x = x + h4 @ Wfc2 + b
    wmma_gemm_kernel<EPI_RESADD_F32><<<dim3(256 / 128, M / 128), 256, 0, stream>>>(
        h4_16, w_fc2[i], fc2_b + i * 256, xbuf, xbuf, nullptr, M, 256, 1024);
  }

  // second output: final x
  copy_f32_kernel<<<(xSz + 255) / 256, 256, 0, stream>>>(xbuf, out + ySz, xSz);

  // downsample conv as im2col GEMM (no bias), then LN -> first output
  im2col_kernel<<<Mds, 256, 0, stream>>>(xbuf, col16);
  wmma_gemm_kernel<EPI_F32><<<dim3(512 / 128, Mds / 128), 256, 0, stream>>>(
      col16, w_ds, nullptr, nullptr, ypre, nullptr, Mds, 512, 768);
  ln_kernel<<<Mds, 512, 0, stream>>>(ypre, dsln_w, dsln_b, nullptr, out, 512);
}
